// CLIPVisionTower_PruneSID_5789615915008
// MI455X (gfx1250) — matrix operand order repaired and checked by reading, not compiled, not run on hardware
//
#include <hip/hip_runtime.h>
#include <math.h>

#define BB 4
#define TT 577
#define NN 576
#define DD 1024
#define HH 16
#define GG 48
#define NEEDC 192

typedef _Float16 half8 __attribute__((ext_vector_type(8)));
typedef _Float16 v16h  __attribute__((ext_vector_type(16)));
typedef float    v8f   __attribute__((ext_vector_type(8)));

#ifndef __has_builtin
#define __has_builtin(x) 0
#endif
#if __has_builtin(__builtin_amdgcn_tensor_load_to_lds)
#define HAVE_TDM 1
#else
#define HAVE_TDM 0
#endif
#if defined(__has_include)
#if __has_include(<hip/amd_detail/amd_gfx1250_TDM.h>)
#define HAVE_TDM_6ARG 1
#else
#define HAVE_TDM_6ARG 0
#endif
#else
#define HAVE_TDM_6ARG 0
#endif

#if HAVE_TDM
typedef unsigned int u32x4t __attribute__((ext_vector_type(4)));
typedef int          i32x4t __attribute__((ext_vector_type(4)));
typedef int          i32x8t __attribute__((ext_vector_type(8)));

// Stage a 16x1024 f16 panel (row-major, stride 1024) into LDS via the TDM.
// D# packing per CDNA5 ISA (2-D tensor, groups 2/3 zero).
__device__ __forceinline__ void tdm_load_panel(const _Float16* gsrc, _Float16* lds_dst) {
  unsigned long long ga = (unsigned long long)(size_t)gsrc;
  unsigned lo = (unsigned)(size_t)(void*)lds_dst;   // flat shared addr: low 32b = LDS offset
  u32x4t g0;
  g0[0] = 1u;                                                      // count=1 (valid D#)
  g0[1] = lo;                                                      // lds_addr
  g0[2] = (unsigned)(ga & 0xFFFFFFFFull);                          // global_addr[31:0]
  g0[3] = (unsigned)((ga >> 32) & 0x01FFFFFFull) | (2u << 30);     // global_addr[56:32] | type=2
  i32x8t g1;
  g1[0] = 0x00010000;            // data_size = 1 -> 2-byte elements; no pad/iterate/multicast
  g1[1] = (int)(DD << 16);       // tensor_dim0 = 1024   (bits 79:48, low half)
  g1[2] = (int)(NN << 16);       // tensor_dim0 hi = 0 | tensor_dim1 = 576 (low half)
  g1[3] = (int)(DD << 16);       // tensor_dim1 hi = 0 | tile_dim0 = 1024
  g1[4] = 16;                    // tile_dim1 = 16 rows; tile_dim2 = 0
  g1[5] = DD;                    // tensor_dim0_stride = 1024 elements (bits 191:160)
  g1[6] = 0;                     // dim0_stride hi | dim1_stride low
  g1[7] = 0;                     // dim1_stride hi
  i32x4t z4 = {0, 0, 0, 0};
#if HAVE_TDM_6ARG
  i32x8t z8 = {0, 0, 0, 0, 0, 0, 0, 0};
  __builtin_amdgcn_tensor_load_to_lds(g0, g1, z4, z4, z8, 0);
#else
  __builtin_amdgcn_tensor_load_to_lds(g0, g1, z4, z4, 0);
#endif
}
#endif  // HAVE_TDM

// ---------------------------------------------------------------------------
// C[b] = X[b] * X[b]^T   (M=N=576, K=1024), f16 inputs, f32 accumulate.
// One block per (b, tm) row-panel: A panel (16x1024, 32 KB) staged to LDS once
// via TDM; 8 waves sweep tn tiles, A fragments from LDS (ds_load_b128),
// B streamed from global, v_wmma_f32_16x16x32_f16 over K in steps of 32.
// ---------------------------------------------------------------------------
__global__ __launch_bounds__(256) void gemm_xxt_wmma(const _Float16* __restrict__ X,
                                                     float* __restrict__ C) {
  __shared__ __align__(16) _Float16 As[16 * DD];     // 32 KB of the 320 KB/WGP LDS
  const int TIL = NN / 16;                           // 36
  int b = blockIdx.x / TIL, tm = blockIdx.x % TIL;
  const _Float16* Xb = X + (size_t)b * NN * DD;
  const _Float16* Ap = Xb + (size_t)(tm * 16) * DD;
  int wv = threadIdx.x >> 5, lane = threadIdx.x & 31;

#if HAVE_TDM
  if (wv == 0) {                                     // wave-uniform; TDM ignores EXEC
    tdm_load_panel(Ap, As);
    __builtin_amdgcn_s_wait_tensorcnt(0);
  }
#else
  for (int i = threadIdx.x; i < 16 * DD / 8; i += 256)
    ((half8*)As)[i] = ((const half8*)Ap)[i];
#endif
  __syncthreads();

  int r16 = lane & 15, hi = lane >> 4;
  const _Float16* ArL = As + (size_t)r16 * DD;       // A row (M = r16) from LDS
  float* Cb = C + (size_t)b * NN * NN;

  for (int tn = wv; tn < TIL; tn += 8) {             // wave-uniform loop: EXEC all-1s
    const _Float16* Br = Xb + (size_t)(tn * 16 + r16) * DD;   // B col (N = r16) = X row
    v8f acc = {0.f, 0.f, 0.f, 0.f, 0.f, 0.f, 0.f, 0.f};
    for (int k = 0; k < DD; k += 32) {
      __builtin_prefetch(Br + k + 128, 0, 3);        // -> global_prefetch_b8
      // 16-bit A 16x32 layout: lane<16 holds K {0..7,16..23}; lane>=16 {8..15,24..31}
      const half8 a0 = *(const half8*)(ArL + k + hi * 8);
      const half8 a1 = *(const half8*)(ArL + k + 16 + hi * 8);
      // 16-bit B 32x16 layout: lane<16 holds K {0..7,16..23}; lane>=16 {8..15,24..31}
      const half8 b0 = *(const half8*)(Br + k + hi * 8);
      const half8 b1 = *(const half8*)(Br + k + 16 + hi * 8);
      v16h av, bv;
#pragma unroll
      for (int i = 0; i < 8; ++i) { av[i] = a0[i]; av[8 + i] = a1[i];
                                    bv[i] = b0[i]; bv[8 + i] = b1[i]; }
      acc = __builtin_amdgcn_wmma_f32_16x16x32_f16(false, av, false, bv,
                                                   (short)0, acc, false, false);
    }
    // C/D layout: VGPR r -> M = r (lanes 0-15) / r+8 (lanes 16-31); N = lane&15
    int col = tn * 16 + r16;
#pragma unroll
    for (int r = 0; r < 8; ++r)
      Cb[(size_t)(tm * 16 + hi * 8 + r) * NN + col] = acc[r];
  }
}

// ---------------------------------------------------------------------------
// Per token: sigmoid-centered features (Mc row) + L2-normalized features (Xn)
// ---------------------------------------------------------------------------
__global__ __launch_bounds__(256) void prep_kernel(const float* __restrict__ hs,
                                                   _Float16* __restrict__ Xn,
                                                   _Float16* __restrict__ Mc) {
  int b = blockIdx.x / NN, n = blockIdx.x % NN, t = threadIdx.x;
  const float4* row = (const float4*)(hs + ((size_t)b * TT + n + 1) * DD);
  float4 x = row[t];
  float s0 = 1.f / (1.f + __expf(-x.x));
  float s1 = 1.f / (1.f + __expf(-x.y));
  float s2 = 1.f / (1.f + __expf(-x.z));
  float s3 = 1.f / (1.f + __expf(-x.w));
  __shared__ float r0[256], r1[256];
  r0[t] = s0 + s1 + s2 + s3;
  r1[t] = x.x * x.x + x.y * x.y + x.z * x.z + x.w * x.w;
  __syncthreads();
  for (int s = 128; s > 0; s >>= 1) {
    if (t < s) { r0[t] += r0[t + s]; r1[t] += r1[t + s]; }
    __syncthreads();
  }
  float mean = r0[0] * (1.f / (float)DD);
  float rn = 1.f / fmaxf(sqrtf(r1[0]), 1e-12f);
  _Float16* xr = Xn + ((size_t)b * NN + n) * DD + t * 4;
  _Float16* mr = Mc + ((size_t)b * NN + n) * DD + t * 4;
  xr[0] = (_Float16)(x.x * rn); xr[1] = (_Float16)(x.y * rn);
  xr[2] = (_Float16)(x.z * rn); xr[3] = (_Float16)(x.w * rn);
  mr[0] = (_Float16)(s0 - mean); mr[1] = (_Float16)(s1 - mean);
  mr[2] = (_Float16)(s2 - mean); mr[3] = (_Float16)(s3 - mean);
}

__global__ void zero_kernel(float* p, int n) {
  int i = blockIdx.x * blockDim.x + threadIdx.x;
  if (i < n) p[i] = 0.f;
}

__global__ void cls_sum_kernel(const float* __restrict__ attn, float* __restrict__ cls) {
  int i = blockIdx.x * blockDim.x + threadIdx.x;
  if (i >= BB * NN) return;
  int b = i / NN, n = i % NN;
  float s = 0.f;
  for (int h = 0; h < HH; ++h)
    s += attn[(((size_t)b * HH + h) * TT + 0) * TT + n + 1];
  cls[i] = s;
}

__global__ __launch_bounds__(256) void simmean_kernel(const float* __restrict__ S,
                                                      float* __restrict__ acc) {
  int b = blockIdx.x / NN, r = blockIdx.x % NN, t = threadIdx.x;
  const float* row = S + ((size_t)b * NN + r) * NN;
  float s = 0.f;
  for (int c = r + 1 + t; c < NN; c += 256) s += row[c];
  __shared__ float red[256];
  red[t] = s; __syncthreads();
  for (int k = 128; k > 0; k >>= 1) { if (t < k) red[t] += red[t + k]; __syncthreads(); }
  if (t == 0 && red[0] != 0.f) atomicAdd(&acc[b], red[0]);
}

// ---- block power iteration for top-48 eigenvectors of G (replaces SVD) ----
__global__ void pinit_kernel(const float* __restrict__ G, float* __restrict__ V) {
  int i = blockIdx.x * blockDim.x + threadIdx.x;
  if (i >= BB * NN * GG) return;
  int b = i / (NN * GG), rem = i % (NN * GG), n = rem / GG, q = rem % GG;
  V[i] = G[((size_t)b * NN + n) * NN + ((q * 37 + 11) % NN)];
}

__global__ void matgv_kernel(const float* __restrict__ G, const float* __restrict__ V,
                             float* __restrict__ V2) {
  int i = blockIdx.x * blockDim.x + threadIdx.x;
  if (i >= BB * NN * GG) return;
  int b = i / (NN * GG), rem = i % (NN * GG), n = rem / GG, q = rem % GG;
  const float* Gr = G + ((size_t)b * NN + n) * NN;
  const float* Vb = V + (size_t)b * NN * GG;
  float s = 0.f;
  for (int j = 0; j < NN; ++j) s += Gr[j] * Vb[(size_t)j * GG + q];
  V2[i] = s;
}

__global__ __launch_bounds__(576) void gs_kernel(const float* __restrict__ Vin,
                                                 float* __restrict__ Vout) {
  __shared__ float red[NN];
  int b = blockIdx.x, t = threadIdx.x;
  const float* vi = Vin + (size_t)b * NN * GG;
  float* vo = Vout + (size_t)b * NN * GG;
  for (int q = 0; q < GG; ++q) vo[(size_t)t * GG + q] = vi[(size_t)t * GG + q];
  __syncthreads();
  for (int q = 0; q < GG; ++q) {
    for (int p = 0; p < q; ++p) {
      red[t] = vo[(size_t)t * GG + q] * vo[(size_t)t * GG + p];
      __syncthreads();
      for (int s = 512; s > 0; s >>= 1) {
        if (t < s && t + s < NN) red[t] += red[t + s];
        __syncthreads();
      }
      float d = red[0];
      vo[(size_t)t * GG + q] -= d * vo[(size_t)t * GG + p];
      __syncthreads();
    }
    red[t] = vo[(size_t)t * GG + q] * vo[(size_t)t * GG + q];
    __syncthreads();
    for (int s = 512; s > 0; s >>= 1) {
      if (t < s && t + s < NN) red[t] += red[t + s];
      __syncthreads();
    }
    float rn = rsqrtf(fmaxf(red[0], 1e-20f));
    vo[(size_t)t * GG + q] *= rn;
    __syncthreads();
  }
}

__global__ void belong_kernel(const float* __restrict__ V, int* __restrict__ belong) {
  int i = blockIdx.x * blockDim.x + threadIdx.x;
  if (i >= BB * NN) return;
  const float* vr = V + (size_t)i * GG;
  float bv = -1.f; int bq = 0;
  for (int q = 0; q < GG; ++q) {
    float a = fabsf(vr[q]);
    if (a > bv) { bv = a; bq = q; }
  }
  belong[i] = bq;
}

// ---- per-(batch,group) sequential similarity NMS ----
__global__ __launch_bounds__(256) void nms_kernel(const float* __restrict__ S,
                                                  const float* __restrict__ cls,
                                                  const int* __restrict__ belong,
                                                  const float* __restrict__ simacc,
                                                  float* __restrict__ keys,
                                                  int* __restrict__ keepc,
                                                  int* __restrict__ groupc) {
  int b = blockIdx.x / GG, g = blockIdx.x % GG, t = threadIdx.x;
  __shared__ float ns[NN];
  __shared__ float rv[256];
  __shared__ int ri[256];
  __shared__ int s_idx;
  const float* Sb = S + (size_t)b * NN * NN;
  const float* cs = cls + b * NN;
  const int* bl = belong + b * NN;
  float* key = keys + ((size_t)b * GG + g) * NN;

  int cnt = 0;
  for (int n = t; n < NN; n += 256) {
    bool in = (bl[n] == g);
    float v = in ? cs[n] : 0.f;
    ns[n] = v; key[n] = v;
    cnt += in ? 1 : 0;
  }
  ri[t] = cnt; __syncthreads();
  for (int s = 128; s > 0; s >>= 1) { if (t < s) ri[t] += ri[t + s]; __syncthreads(); }
  if (t == 0) groupc[blockIdx.x] = ri[0];
  const float thr = 6.f * simacc[b] / (0.5f * (float)NN * (float)(NN - 1));
  __syncthreads();

  float gs = 1000.f;
  int count = 0;
  while (true) {
    float bv = 0.f; int bidx = -1;
    for (int n = t; n < NN; n += 256) {
      float v = ns[n];
      if (v > bv) { bv = v; bidx = n; }
    }
    rv[t] = bv; ri[t] = bidx; __syncthreads();
    for (int s = 128; s > 0; s >>= 1) {
      if (t < s) {
        bool sw = (rv[t + s] > rv[t]) ||
                  (rv[t + s] == rv[t] && ri[t + s] >= 0 && (ri[t] < 0 || ri[t + s] < ri[t]));
        if (sw) { rv[t] = rv[t + s]; ri[t] = ri[t + s]; }
      }
      __syncthreads();
    }
    if (t == 0) s_idx = (rv[0] > 0.f) ? ri[0] : -1;
    __syncthreads();
    int idx = s_idx;
    if (idx < 0) break;
    if (t == 0) { key[idx] = gs; ns[idx] = 0.f; }
    __syncthreads();
    const float* Srow = Sb + (size_t)idx * NN;
    for (int n = t; n < NN; n += 256)
      if (ns[n] > 0.f && Srow[n] > thr) ns[n] = 0.f;
    count++; gs -= 1.f;
    __syncthreads();
  }
  if (t == 0) keepc[blockIdx.x] = count;
}

// stable descending argsort of keys via rank counting
__global__ __launch_bounds__(256) void ranksort_kernel(const float* __restrict__ keys,
                                                       int* __restrict__ sortidx) {
  int bg = blockIdx.x, t = threadIdx.x;
  __shared__ float k[NN];
  const float* src = keys + (size_t)bg * NN;
  for (int n = t; n < NN; n += 256) k[n] = src[n];
  __syncthreads();
  int* dst = sortidx + (size_t)bg * NN;
  for (int n = t; n < NN; n += 256) {
    float kn = k[n];
    int r = 0;
    for (int j = 0; j < NN; ++j) {
      float kj = k[j];
      r += (kj > kn) || (kj == kn && j < n);
    }
    dst[r] = n;
  }
}

// token-budget allocation (1 thread / batch — 48-element logic)
__global__ void alloc_kernel(const int* __restrict__ keepc, const int* __restrict__ groupc,
                             int* __restrict__ tokend, int* __restrict__ offs) {
  if (threadIdx.x != 0) return;
  int b = blockIdx.x;
  const int* kc = keepc + b * GG;
  const int* gc = groupc + b * GG;
  int lower[GG], upper[GG], d[GG], sidx[GG];
  int lowsum = 0;
  for (int g = 0; g < GG; ++g) { lower[g] = gc[g] < 1 ? gc[g] : 1; lowsum += lower[g]; }
  int us = 0;
  for (int g = 0; g < GG; ++g) {
    int u = 15 < gc[g] ? 15 : gc[g];           // 5*ceil(192/64)=15
    if (kc[g] < u) u = kc[g];
    upper[g] = u; us += u;
  }
  int guard = 0;
  while (us < NEEDC && guard++ < 1024) {
    us = 0;
    for (int g = 0; g < GG; ++g) {
      int u = upper[g] + 1; if (u > gc[g]) u = gc[g];
      upper[g] = u; us += u;
    }
  }
  int other = NEEDC - lowsum - 1; if (other < 0) other = 0;
  float ksum = 0.f;
  for (int g = 0; g < GG; ++g) ksum += (float)kc[g];
  if (ksum <= 0.f) ksum = 1.f;
  float csum = 0.f; int prev = 0;
  for (int g = 0; g < GG; ++g) {
    csum += (float)kc[g] / ksum;
    int cr = (int)rintf(csum * (float)other);
    int od = cr - prev; prev = cr;
    int td = od + lower[g];
    if (td > upper[g]) td = upper[g];
    if (td < 0) td = 0;
    d[g] = td;
  }
  for (int g = 0; g < GG; ++g) sidx[g] = g;
  for (int a = 1; a < GG; ++a) {               // stable insertion sort by -keepc
    int key = sidx[a], kv = kc[key], j = a - 1;
    while (j >= 0 && kc[sidx[j]] < kv) { sidx[j + 1] = sidx[j]; j--; }
    sidx[j + 1] = key;
  }
  int target = other + lowsum;
  int ds = 0;
  for (int g = 0; g < GG; ++g) ds += d[g];
  int fg = 0; guard = 0;
  while (ds < target && guard++ < 4096) {
    int g2 = sidx[fg < GG - 1 ? fg : GG - 1];
    int fill = upper[g2] - d[g2];
    int remn = target - ds;
    if (fill > remn) fill = remn;
    if (fill < 0) fill = 0;
    d[g2] += fill; ds += fill; fg++;
  }
  int off = 0;
  for (int g = 0; g < GG; ++g) { tokend[b * GG + g] = d[g]; offs[b * GG + g] = off; off += d[g]; }
}

__global__ void gather_idx_kernel(const int* __restrict__ sortidx,
                                  const int* __restrict__ tokend,
                                  const int* __restrict__ offs,
                                  int* __restrict__ bi, float* __restrict__ out_idx) {
  if (threadIdx.x != 0) return;
  int b = blockIdx.x;
  for (int i = 0; i < NEEDC; ++i) bi[b * NEEDC + i] = 0;      // pos 0 = CLS token
  for (int g = 0; g < GG; ++g) {
    int td = tokend[b * GG + g], off = offs[b * GG + g];
    for (int j = 0; j < td; ++j) {
      int pos = off + j + 1;
      if (pos < NEEDC)
        bi[b * NEEDC + pos] = sortidx[((size_t)(b * GG + g)) * NN + j] + 1;
    }
  }
  for (int i = 0; i < NEEDC; ++i) out_idx[b * NEEDC + i] = (float)bi[b * NEEDC + i];
}

__global__ __launch_bounds__(256) void out_gather_kernel(const float* __restrict__ hs,
                                                         const int* __restrict__ bi,
                                                         float* __restrict__ out) {
  int b = blockIdx.x / NEEDC, i = blockIdx.x % NEEDC, t = threadIdx.x;
  int tok = bi[b * NEEDC + i];
  const float4* src = (const float4*)(hs + ((size_t)b * TT + tok) * DD);
  float4* dst = (float4*)(out + ((size_t)b * NEEDC + i) * DD);
  dst[t] = src[t];
}

// ---------------------------------------------------------------------------
extern "C" void kernel_launch(void* const* d_in, const int* in_sizes, int n_in,
                              void* d_out, int out_size, void* d_ws, size_t ws_size,
                              hipStream_t stream) {
  (void)in_sizes; (void)n_in; (void)out_size; (void)ws_size;
  const float* hs   = (const float*)d_in[0];
  const float* attn = (const float*)d_in[1];
  float* out = (float*)d_out;
  float* out_idx = out + (size_t)BB * NEEDC * DD;

  char* w = (char*)d_ws;
  auto take = [&](size_t bytes) -> char* {
    char* p = w;
    w += (bytes + 255) & ~(size_t)255;
    return p;
  };
  _Float16* Xn  = (_Float16*)take((size_t)BB * NN * DD * 2);
  _Float16* Mc  = (_Float16*)take((size_t)BB * NN * DD * 2);
  float* S      = (float*)take((size_t)BB * NN * NN * 4);
  float* G      = (float*)take((size_t)BB * NN * NN * 4);
  float* V      = (float*)take((size_t)BB * NN * GG * 4);
  float* V2     = (float*)take((size_t)BB * NN * GG * 4);
  float* cls    = (float*)take((size_t)BB * NN * 4);
  float* simacc = (float*)take((size_t)BB * 4);
  int* belong   = (int*)take((size_t)BB * NN * 4);
  float* keys   = (float*)take((size_t)BB * GG * NN * 4);
  int* keepc    = (int*)take((size_t)BB * GG * 4);
  int* groupc   = (int*)take((size_t)BB * GG * 4);
  int* tokend   = (int*)take((size_t)BB * GG * 4);
  int* offs     = (int*)take((size_t)BB * GG * 4);
  int* sortidx  = (int*)take((size_t)BB * GG * NN * 4);
  int* bi       = (int*)take((size_t)BB * NEEDC * 4);

  zero_kernel<<<1, 32, 0, stream>>>(simacc, BB);
  prep_kernel<<<BB * NN, 256, 0, stream>>>(hs, Xn, Mc);
  cls_sum_kernel<<<(BB * NN + 255) / 256, 256, 0, stream>>>(attn, cls);

  // one block per (batch, 16-row panel): 4*36 = 144 blocks, 8 waves each
  gemm_xxt_wmma<<<BB * (NN / 16), 256, 0, stream>>>(Xn, S);   // similarity
  gemm_xxt_wmma<<<BB * (NN / 16), 256, 0, stream>>>(Mc, G);   // Gram (for PCA)

  simmean_kernel<<<BB * NN, 256, 0, stream>>>(S, simacc);

  pinit_kernel<<<(BB * NN * GG + 255) / 256, 256, 0, stream>>>(G, V);
  for (int it = 0; it < 4; ++it) {
    matgv_kernel<<<(BB * NN * GG + 255) / 256, 256, 0, stream>>>(G, V, V2);
    gs_kernel<<<BB, NN, 0, stream>>>(V2, V);
  }
  belong_kernel<<<(BB * NN + 255) / 256, 256, 0, stream>>>(V, belong);

  nms_kernel<<<BB * GG, 256, 0, stream>>>(S, cls, belong, simacc, keys, keepc, groupc);
  ranksort_kernel<<<BB * GG, 256, 0, stream>>>(keys, sortidx);
  alloc_kernel<<<BB, 32, 0, stream>>>(keepc, groupc, tokend, offs);
  gather_idx_kernel<<<BB, 32, 0, stream>>>(sortidx, tokend, offs, bi, out_idx);
  out_gather_kernel<<<BB * NEEDC, 256, 0, stream>>>(hs, bi, out);
}
